// CrossConvLayer_51170240364937
// MI455X (gfx1250) — matrix operand order repaired and checked by reading, not compiled, not run on hardware
//
#include <hip/hip_runtime.h>
#include <hip/hip_bf16.h>

typedef __bf16 bf16_t;
typedef bf16_t v16bf __attribute__((ext_vector_type(16)));
typedef float  v8f   __attribute__((ext_vector_type(8)));

#define NB    4
#define NPTS  4096
#define LEAKY 0.1f

// ---------------------------------------------------------------------------
// KNN: for each query point, 16 nearest targets (squared distance, ascending)
// ---------------------------------------------------------------------------
__global__ __launch_bounds__(256) void knn_kernel(
    const float* __restrict__ pcq,   // [B,3,N] query coords
    const float* __restrict__ pct,   // [B,3,N] target coords
    int* __restrict__ idx_out)       // [B,N,16]
{
  const int b  = blockIdx.y;
  const int pt = blockIdx.x * 256 + threadIdx.x;
  __shared__ float sx[1024], sy[1024], sz[1024];
  const float qx = pcq[(b*3+0)*NPTS + pt];
  const float qy = pcq[(b*3+1)*NPTS + pt];
  const float qz = pcq[(b*3+2)*NPTS + pt];
  float bd[16]; int bi[16];
  #pragma unroll
  for (int r=0;r<16;++r){ bd[r]=3.0e38f; bi[r]=0; }
  for (int t0=0;t0<NPTS;t0+=1024){
    __syncthreads();
    for (int j=threadIdx.x;j<1024;j+=256){
      sx[j]=pct[(b*3+0)*NPTS+t0+j];
      sy[j]=pct[(b*3+1)*NPTS+t0+j];
      sz[j]=pct[(b*3+2)*NPTS+t0+j];
    }
    __syncthreads();
    for (int j=0;j<1024;++j){
      const float dx=sx[j]-qx, dy=sy[j]-qy, dz=sz[j]-qz;
      float cd = dx*dx+dy*dy+dz*dz;
      if (cd < bd[15]) {
        int ci = t0+j;
        #pragma unroll
        for (int r=0;r<16;++r){
          if (cd < bd[r]) {
            float td=bd[r]; int ti=bi[r];
            bd[r]=cd; bi[r]=ci; cd=td; ci=ti;
          }
        }
      }
    }
  }
  #pragma unroll
  for (int r=0;r<16;++r) idx_out[((size_t)(b*NPTS)+pt)*16 + r] = bi[r];
}

// ---------------------------------------------------------------------------
// [B,C,N] -> [B,N,C]   (point-major copy for fast neighbor gathers)
// ---------------------------------------------------------------------------
__global__ __launch_bounds__(256) void transpose_feat_kernel(
    const float* __restrict__ f, float* __restrict__ fT, int C, int total)
{
  const int t = blockIdx.x*256 + threadIdx.x;
  if (t >= total) return;
  const int n = t % NPTS;
  const int c = (t / NPTS) % C;
  const int b = t / (NPTS * C);
  fT[((size_t)b*NPTS + n)*C + c] = f[t];
}

// ---------------------------------------------------------------------------
// lin_w [K,128] f32 -> lwT [128,K] bf16  (B-operand friendly layout)
// ---------------------------------------------------------------------------
__global__ __launch_bounds__(256) void prep_w_kernel(
    const float* __restrict__ lw, bf16_t* __restrict__ lwT, int K, int total)
{
  const int t = blockIdx.x*256 + threadIdx.x;
  if (t >= total) return;
  const int o = t % 128;
  const int k = t / 128;
  lwT[(size_t)o*K + k] = (bf16_t)lw[t];
}

// ---------------------------------------------------------------------------
// Fused cross layer. One WG (256 thr = 8 waves) = 16 points.
//   stage A (VALU): WeightNet per (point,k) -> wtsT[p][w][k] bf16
//   per 16-wide d block:
//     phase1: gather neighbor features -> ptsT[p][d][k] bf16 (or own-feat qS)
//     phase2: agg via WMMA (16x16x32, K zero-padded 16->32), 2 points/wave,
//             D tile -> aggS[p][d*16+w] bf16  (A-striping of the big GEMM)
//     phase3: big GEMM: wave j owns output cols [16j,16j+16), 8 WMMA per block
// ---------------------------------------------------------------------------
template<int D1, int D2>
__global__ __launch_bounds__(256) void cross_kernel(
    const float* __restrict__ pcq,        // [B,3,N] query coords
    const float* __restrict__ pct,        // [B,3,N] target coords
    const int*   __restrict__ knn_idx,    // [B,N,16]
    const float* __restrict__ featq_cm,   // [B,D1,N] own-point feats (channel-major)
    const float* __restrict__ featn_pm,   // [B,N,D2] neighbor feats (point-major)
    const float* __restrict__ w0, const float* __restrict__ b0,
    const float* __restrict__ w1, const float* __restrict__ b1,
    const float* __restrict__ w2, const float* __restrict__ b2,
    const bf16_t* __restrict__ lwT,       // [128,(D1+D2)*16] bf16
    const float* __restrict__ lb,         // [128]
    float* __restrict__ out_cm,           // [B,128,N]
    float* __restrict__ out_pm)           // [B,N,128] or nullptr
{
  constexpr int D    = D1 + D2;
  constexpr int KTOT = D * 16;
  constexpr int NDB  = D / 16;     // 16-wide d blocks
  constexpr int FDB  = D1 / 16;    // broadcast (own-feature) blocks

  __shared__ int    idxS[16][16];        // [p][k]
  __shared__ bf16_t wtsT[16][16][24];    // [p][w][k] (padded rows, 16 valid k)
  __shared__ bf16_t ptsT[16][16][24];    // [p][d][k] (padded rows, 16 valid k)
  __shared__ float  qS[16][16];          // [p][dd]
  __shared__ bf16_t aggS[16][264];       // [p][dd*16+w], padded row

  const int tid  = threadIdx.x;
  const int p    = tid >> 4;       // point in tile
  const int w    = tid & 15;       // k (stage A / gathers) or w column
  const int lane = tid & 31;
  const int ln   = lane & 15;
  const int hl   = lane >> 4;
  const int wave = tid >> 5;
  const int b    = blockIdx.y;
  const int tile = blockIdx.x;
  const int pt   = tile*16 + p;

  // ---- stage A: neighbor gather + WeightNet (lane handles k = w) ----
  const int nb = knn_idx[((size_t)(b*NPTS)+pt)*16 + w];
  idxS[p][w] = nb;
  const float qx = pcq[(b*3+0)*NPTS+pt];
  const float qy = pcq[(b*3+1)*NPTS+pt];
  const float qz = pcq[(b*3+2)*NPTS+pt];
  const float dx = pct[(b*3+0)*NPTS+nb]-qx;
  const float dy = pct[(b*3+1)*NPTS+nb]-qy;
  const float dz = pct[(b*3+2)*NPTS+nb]-qz;
  float h0[8], h1[8];
  #pragma unroll
  for (int o=0;o<8;++o){
    float s = dx*w0[o] + dy*w0[8+o] + dz*w0[16+o] + b0[o];
    h0[o] = fmaxf(s, 0.f);
  }
  #pragma unroll
  for (int o=0;o<8;++o){
    float s = b1[o];
    #pragma unroll
    for (int i=0;i<8;++i) s += h0[i]*w1[i*8+o];
    h1[o] = fmaxf(s, 0.f);
  }
  #pragma unroll
  for (int o=0;o<16;++o){
    float s = b2[o];
    #pragma unroll
    for (int i=0;i<8;++i) s += h1[i]*w2[i*16+o];
    wtsT[p][o][w] = (bf16_t)fmaxf(s, 0.f);   // transposed store: [p][w][k]
  }
  __syncthreads();
  // column sum over k (for the broadcast own-feature half: rank-1 agg)
  float wsum = 0.f;
  #pragma unroll
  for (int k=0;k<16;++k) wsum += (float)wtsT[p][w][k];

  // ---- main loop over d blocks ----
  v8f c = {0,0,0,0,0,0,0,0};
  const int ncol = wave*16 + ln;                          // output channel
  const bf16_t* __restrict__ lwRow = lwT + (size_t)ncol * KTOT;
  const bf16_t* __restrict__ arow  = &aggS[ln][0];        // big-GEMM A row

  for (int db=0; db<NDB; ++db) {
    // ---- phase 1: loads ----
    if (db < FDB) {
      qS[p][w] = featq_cm[((size_t)(b*D1) + (db*16 + w))*NPTS + pt];
    } else {
      const int dloc = (db - FDB)*16;
      const float4* src = (const float4*)(featn_pm +
                          ((size_t)b*NPTS + idxS[p][w])*D2 + dloc);
      const float4 f0=src[0], f1=src[1], f2=src[2], f3=src[3];
      const float v[16] = {f0.x,f0.y,f0.z,f0.w, f1.x,f1.y,f1.z,f1.w,
                           f2.x,f2.y,f2.z,f2.w, f3.x,f3.y,f3.z,f3.w};
      #pragma unroll
      for (int d=0; d<16; ++d) ptsT[p][d][w] = (bf16_t)v[d];  // k = w
    }
    __syncthreads();

    // ---- phase 2: agg -> aggS ----
    if (db < FDB) {
      #pragma unroll
      for (int dd=0; dd<16; ++dd)
        aggS[p][dd*16+w] = (bf16_t)(qS[p][dd] * wsum);
    } else {
      // agg[d][w] = sum_k pts[k][d]*wts[k][w]; one 16x16x32 WMMA per point,
      // K 16..31 zero-padded on BOTH operands (avoid NaN poisoning).
      #pragma unroll
      for (int pp2=0; pp2<2; ++pp2) {
        const int pp = wave*2 + pp2;
        v16bf A, Bm;
        const uint4 z = make_uint4(0u,0u,0u,0u);
        // A: lane M=d=ln, elems j=0..7 -> K=8*hl+j ; elems 8..15 -> K>=16 = 0
        ((uint4*)&A)[0] = *(const uint4*)&ptsT[pp][ln][8*hl];
        ((uint4*)&A)[1] = z;
        // B: lane N=w=ln, elems j=0..7 -> K=8*hl+j ; elems 8..15 -> K>=16 = 0
        uint4 b0v = *(const uint4*)&wtsT[pp][ln][8*hl], b1v = z;
        ((uint4*)&Bm)[0] = b0v;
        ((uint4*)&Bm)[1] = b1v;
        v8f ca = {0,0,0,0,0,0,0,0};
        ca = __builtin_amdgcn_wmma_f32_16x16x32_bf16(
                 false, A, false, Bm, (short)0, ca, false, false);
        // D: lane col n=w=ln, elem r -> row d = r+8*hl ; scatter into aggS
        #pragma unroll
        for (int r=0;r<8;++r)
          aggS[pp][(r+8*hl)*16 + ln] = (bf16_t)ca[r];
      }
    }
    __syncthreads();

    // ---- phase 3: big GEMM, 256-wide K chunk = 8 x (16x16x32 bf16 WMMA) ----
    #pragma unroll
    for (int sub=0; sub<8; ++sub) {
      const int kc = sub*32;
      v16bf A, Bm;
      // A (16x32): lane M=point=ln; runs K=kc+8*hl.. and kc+16+8*hl..
      ((uint4*)&A)[0] = *(const uint4*)(arow + kc + 8*hl);
      ((uint4*)&A)[1] = *(const uint4*)(arow + kc + 16 + 8*hl);
      // B (32x16): lane N=ln; runs K=kc+8*hl.. and kc+16+8*hl..
      const bf16_t* bp = lwRow + db*256 + kc + 8*hl;
      ((uint4*)&Bm)[0] = *(const uint4*)(bp);
      ((uint4*)&Bm)[1] = *(const uint4*)(bp + 16);
      c = __builtin_amdgcn_wmma_f32_16x16x32_bf16(
              false, A, false, Bm, (short)0, c, false, false);
    }
  }

  // ---- epilogue: bias + leaky relu, store ----
  const float bias = lb[ncol];
  #pragma unroll
  for (int r=0;r<8;++r){
    float v = c[r] + bias;
    c[r] = v > 0.f ? v : LEAKY*v;
  }
  {
    // D tile: lane col = channel ncol, element r -> point m = r + 8*hl
    float* o = out_cm + ((size_t)(b*128 + ncol))*NPTS + tile*16 + 8*hl;
    float4 lo, hi;
    lo.x=c[0]; lo.y=c[1]; lo.z=c[2]; lo.w=c[3];
    hi.x=c[4]; hi.y=c[5]; hi.z=c[6]; hi.w=c[7];
    *(float4*)(o)   = lo;
    *(float4*)(o+4) = hi;
  }
  if (out_pm != nullptr) {
    #pragma unroll
    for (int r=0;r<8;++r)
      out_pm[((size_t)(b*NPTS) + tile*16 + 8*hl + r)*128 + ncol] = c[r];
  }
}

// ---------------------------------------------------------------------------
extern "C" void kernel_launch(void* const* d_in, const int* in_sizes, int n_in,
                              void* d_out, int out_size, void* d_ws, size_t ws_size,
                              hipStream_t stream)
{
  (void)in_sizes; (void)n_in; (void)out_size; (void)ws_size;
  const float* pc1    = (const float*)d_in[0];
  const float* pc2    = (const float*)d_in[1];
  const float* feat1  = (const float*)d_in[2];
  const float* feat2  = (const float*)d_in[3];
  const float* wn1_w0 = (const float*)d_in[4];
  const float* wn1_b0 = (const float*)d_in[5];
  const float* wn1_w1 = (const float*)d_in[6];
  const float* wn1_b1 = (const float*)d_in[7];
  const float* wn1_w2 = (const float*)d_in[8];
  const float* wn1_b2 = (const float*)d_in[9];
  const float* lin1_w = (const float*)d_in[10];
  const float* lin1_b = (const float*)d_in[11];
  const float* wn2_w0 = (const float*)d_in[12];
  const float* wn2_b0 = (const float*)d_in[13];
  const float* wn2_w1 = (const float*)d_in[14];
  const float* wn2_b1 = (const float*)d_in[15];
  const float* wn2_w2 = (const float*)d_in[16];
  const float* wn2_b2 = (const float*)d_in[17];
  const float* lin2_w = (const float*)d_in[18];
  const float* lin2_b = (const float*)d_in[19];

  const size_t BN = (size_t)NB * NPTS;

  // workspace carve (every region fully rewritten each call)
  char* ws = (char*)d_ws; size_t off = 0;
  auto carve = [&](size_t bytes)->char* {
    char* r = ws + off; off = (off + bytes + 255) & ~(size_t)255; return r;
  };
  int*    idx12 = (int*)   carve(BN*16*sizeof(int));
  int*    idx21 = (int*)   carve(BN*16*sizeof(int));
  float*  fT1   = (float*) carve(BN*64*sizeof(float));
  float*  fT2   = (float*) carve(BN*64*sizeof(float));
  float*  f1nT  = (float*) carve(BN*128*sizeof(float));
  float*  f2nT  = (float*) carve(BN*128*sizeof(float));
  bf16_t* lwT1  = (bf16_t*)carve((size_t)128*2048*sizeof(bf16_t));
  bf16_t* lwT2  = (bf16_t*)carve((size_t)128*4096*sizeof(bf16_t));

  float* out    = (float*)d_out;
  float* f1n_cm = out;                              // [B,128,N]
  float* f2n_cm = out + (size_t)NB*128*NPTS;
  float* f1f_cm = out + (size_t)2*NB*128*NPTS;

  // ---- prep: transposed features + bf16 transposed GEMM weights ----
  {
    const int tot = (int)(BN*64);
    transpose_feat_kernel<<<dim3((tot+255)/256), dim3(256), 0, stream>>>(feat1, fT1, 64, tot);
    transpose_feat_kernel<<<dim3((tot+255)/256), dim3(256), 0, stream>>>(feat2, fT2, 64, tot);
  }
  {
    const int t1 = 2048*128, t2 = 4096*128;
    prep_w_kernel<<<dim3((t1+255)/256), dim3(256), 0, stream>>>(lin1_w, lwT1, 2048, t1);
    prep_w_kernel<<<dim3((t2+255)/256), dim3(256), 0, stream>>>(lin2_w, lwT2, 4096, t2);
  }

  // ---- KNN (idx12 reused by cross1 and cross3) ----
  knn_kernel<<<dim3(NPTS/256, NB), dim3(256), 0, stream>>>(pc1, pc2, idx12);
  knn_kernel<<<dim3(NPTS/256, NB), dim3(256), 0, stream>>>(pc2, pc1, idx21);

  // ---- cross 1: x1 queries, feat1(own) + feat2(neighbors) ----
  cross_kernel<64,64><<<dim3(NPTS/16, NB), dim3(256), 0, stream>>>(
      pc1, pc2, idx12, feat1, fT2,
      wn1_w0, wn1_b0, wn1_w1, wn1_b1, wn1_w2, wn1_b2,
      lwT1, lin1_b, f1n_cm, f1nT);

  // ---- cross 2: x2 queries, feat2(own) + feat1(neighbors), same weights ----
  cross_kernel<64,64><<<dim3(NPTS/16, NB), dim3(256), 0, stream>>>(
      pc2, pc1, idx21, feat2, fT1,
      wn1_w0, wn1_b0, wn1_w1, wn1_b1, wn1_w2, wn1_b2,
      lwT1, lin1_b, f2n_cm, f2nT);

  // ---- cross 3: x1 queries, f1n(own, channel-major) + f2n(neighbors) ----
  cross_kernel<128,128><<<dim3(NPTS/16, NB), dim3(256), 0, stream>>>(
      pc1, pc2, idx12, f1n_cm, f2nT,
      wn2_w0, wn2_b0, wn2_w1, wn2_b1, wn2_w2, wn2_b2,
      lwT2, lin2_b, f1f_cm, nullptr);
}
